// KSplanifoldSpine_46634754899971
// MI455X (gfx1250) — matrix operand order repaired and checked by reading, not compiled, not run on hardware
//
#include <hip/hip_runtime.h>
#include <stdint.h>

// ---------------------------------------------------------------------------
// MI455X (gfx1250) fused "planifold spine" kernel, v3b.
// N=524288 rows, D=128, R=16. HBM floor ~23us @23.3TB/s; GEMM stages run
// v_wmma_f32_16x16x32_f16 (f16 in, f32 accumulate); reductions/elementwise f32.
// v3: (a) v_cvt_pk_rtz_f16_f32 for all f16 pair packing (kills the 16-bit
// insert chains seen in v2), (b) staggered wf/df lifetimes to fit <256 VGPRs.
// v3b: fix cvt_pkrtz return type (__fp16 ext_vector(2)).
// ---------------------------------------------------------------------------

typedef __attribute__((ext_vector_type(16))) _Float16 v16h;
typedef __attribute__((ext_vector_type(8)))  float    v8f;
typedef __attribute__((ext_vector_type(2)))  __fp16   fp16x2;

#define NROWS   524288
#define DDIM    128
#define RRANK   16
#define WAVES   4
#define BLOCK_T (WAVES * 32)        // 128 threads
#define TILE_ROWS (WAVES * 16)      // 64 rows per block-tile
#define NTILES  (NROWS / TILE_ROWS) // 8192
#define GRID    2048                // grid-stride: 4 tiles per block

struct Params {
    const float* r;
    const float* spine;
    const float* P0;
    const float* P1;
    const float* U[6];   // (D,R) row-major
    const float* V[6];   // (R,D) row-major
    float*       out;
};

union FragH { uint32_t u[8]; v16h h; };
union AccF  { v8f v; float f[8]; };

// single-instruction packed f32x2 -> f16x2 (v_cvt_pk_rtz_f16_f32)
static __device__ __forceinline__ uint32_t pack2h(float a, float b) {
    union { fp16x2 h; uint32_t u; } p;
    p.h = __builtin_amdgcn_cvt_pkrtz(a, b);
    return p.u;
}
static __device__ __forceinline__ uint16_t to_h16(float a) {
    union { _Float16 h; uint16_t u; } p;
    p.h = (_Float16)a;
    return p.u;
}
// cross-lane read (wave32): value of 'v' from lane 'src'
static __device__ __forceinline__ float bcastf(float v, int src) {
    return __int_as_float(__builtin_amdgcn_ds_bpermute(src << 2, __float_as_int(v)));
}

__global__ __launch_bounds__(BLOCK_T)
void spine_wmma_kernel(Params p) {
    // B-fragment-ready packed-f16 weight staging:
    //  sV[m][dpair][j]  = (V_m[j][2dp], V_m[j][2dp+1])          -> GEMM1 B
    //  sU[cpair][d]     = (Ucat^T[2cp][d], Ucat^T[2cp+1][d])    -> GEMM2 B
    //  sY[wave][row][c] : per-wave Y tile (16x96 f16, padded row=98)
    //  sSp[hi][pos]     : spine slice in A-fragment column order (broadcast)
    __shared__ uint32_t sV[6][64][16];     // 24 KB
    __shared__ uint32_t sU[48][128];       // 24 KB
    __shared__ uint16_t sY[WAVES][16][98]; // 12.25 KB
    __shared__ float    sSp[2][64];        // 0.5 KB   (total ~62.2 KB)

    const int tid  = threadIdx.x;
    const int wave = tid >> 5;
    const int lane = tid & 31;
    const int l15  = lane & 15;
    const int hi   = lane >> 4;

    // ---- stage weights (f32 -> packed f16) + spine slice, once per block ----
#pragma unroll
    for (int m = 0; m < 6; ++m) {
        const float* Vm = p.V[m];
        for (int idx = tid; idx < 1024; idx += BLOCK_T) {
            int dp = idx >> 4, j = idx & 15;
            sV[m][dp][j] = pack2h(Vm[j * DDIM + 2 * dp], Vm[j * DDIM + 2 * dp + 1]);
        }
        const float* Um = p.U[m];
        for (int idx = tid; idx < 1024; idx += BLOCK_T) {
            int cp8 = idx >> 7;           // 0..7
            int d   = idx & 127;
            int j   = cp8 * 2;            // column pair within this U
            sU[m * 8 + cp8][d] = pack2h(Um[d * RRANK + j], Um[d * RRANK + j + 1]);
        }
    }
    {   // sSp[h][c*8+u] = spine[16c + 8h + u]
        int h = tid >> 6, pos = tid & 63;      // 128 threads -> one element each
        int c = pos >> 3, u = pos & 7;
        sSp[h][pos] = p.spine[16 * c + 8 * h + u];
    }
    __syncthreads();

    // ---- persistent per-lane data ----
    float spn2h = 0.f;
#pragma unroll
    for (int c8 = 0; c8 < 8; ++c8) {
        float4 a = *(const float4*)&sSp[hi][c8 * 8];
        float4 b = *(const float4*)&sSp[hi][c8 * 8 + 4];
        spn2h += a.x*a.x + a.y*a.y + a.z*a.z + a.w*a.w
               + b.x*b.x + b.y*b.y + b.z*b.z + b.w*b.w;
    }
    float spn2 = spn2h + bcastf(spn2h, lane ^ 16);   // full ||spine||^2
    const float inv_spn2 = 1.0f / spn2;

    float P0v[8], P1v[8];
#pragma unroll
    for (int nt = 0; nt < 8; ++nt) {
        P0v[nt] = p.P0[16 * nt + l15];
        P1v[nt] = p.P1[16 * nt + l15];
    }

    const float eps2 = 0.05f * 0.05f;
    const float epsD = eps2 / (float)DDIM;

    for (int tile = blockIdx.x; tile < NTILES; tile += gridDim.x) {
        const int rowBase = tile * TILE_ROWS + wave * 16;
        const int myRow   = rowBase + l15;           // this lane's row (2 lanes/row)
        const float* rrow = p.r + (size_t)myRow * DDIM;

        // prefetch next grid-stride tile's r region for this wave (8KB, 2 lines/lane)
        if (tile + (int)gridDim.x < NTILES) {
            const char* nxt = (const char*)(p.r +
                (size_t)(tile + gridDim.x) * TILE_ROWS * DDIM + (size_t)wave * 16 * DDIM);
            __builtin_prefetch(nxt + lane * 256, 0, 3);
            __builtin_prefetch(nxt + lane * 256 + 128, 0, 3);
        }

        // ---- phase A: load this lane's 64 columns; row reductions ----
        float rr[64];
        float s1 = 0.f, s2 = 0.f, sd = 0.f;
#pragma unroll
        for (int c = 0; c < 8; ++c) {
            float4 a  = *(const float4*)(rrow + 16 * c + 8 * hi);
            float4 b  = *(const float4*)(rrow + 16 * c + 8 * hi + 4);
            float4 qa = *(const float4*)&sSp[hi][c * 8];
            float4 qb = *(const float4*)&sSp[hi][c * 8 + 4];
            float* q  = &rr[c * 8];
            q[0] = a.x; q[1] = a.y; q[2] = a.z; q[3] = a.w;
            q[4] = b.x; q[5] = b.y; q[6] = b.z; q[7] = b.w;
            s1 += a.x + a.y + a.z + a.w + b.x + b.y + b.z + b.w;
            s2 += a.x*a.x + a.y*a.y + a.z*a.z + a.w*a.w
                + b.x*b.x + b.y*b.y + b.z*b.z + b.w*b.w;
            sd += a.x*qa.x + a.y*qa.y + a.z*qa.z + a.w*qa.w
                + b.x*qb.x + b.y*qb.y + b.z*qb.z + b.w*qb.w;
        }
        // lanes L and L^16 hold complementary halves of the same row
        s1 += bcastf(s1, lane ^ 16);
        s2 += bcastf(s2, lane ^ 16);
        sd += bcastf(sd, lane ^ 16);

        const float sigma = s1, rsq = s2, rdots = sd;
        const float t     = 1.0f / (1.0f + __expf(-rdots));
        const float coef  = rdots * inv_spn2;                 // projection coefficient
        float dn2         = fmaxf(rsq - rdots * rdots * inv_spn2, 0.f);
        const float invdt = rsqrtf(1.0f + dn2);               // DELTA_MAX = 1
        const float denom = 1.0f / (sigma * sigma + eps2);
        const float aw    = sigma * denom;
        const float bw    = epsD * denom;
        const float t2 = t * t, t3 = t2 * t;
        const float h00 = 2.f * t3 - 3.f * t2 + 1.f;
        const float h01 = -2.f * t3 + 3.f * t2;
        const float h10 = t3 - 2.f * t2 + t;
        const float h11 = t3 - t2;

        // Hermite coeffs per C-row (row = v + 8*hi) via cross-lane broadcast
        float H00[8], H01[8], H10[8], H11[8];
#pragma unroll
        for (int v = 0; v < 8; ++v) {
            int src = v + 8 * hi;
            H00[v] = bcastf(h00, src); H01[v] = bcastf(h01, src);
            H10[v] = bcastf(h10, src); H11[v] = bcastf(h11, src);
        }

        // ensure previous iteration's sY reads completed before overwrite
        asm volatile("s_wait_dscnt 0" ::: "memory");

        // one 16x16 y tile:  y_m = A(16x128) @ V_m^T(128x16), scaled, -> sY
        auto gemm1_one = [&](const uint32_t* af, int m, const float* Cm) {
            AccF acc;
#pragma unroll
            for (int v = 0; v < 8; ++v) acc.f[v] = 0.f;
#pragma unroll
            for (int kb = 0; kb < 4; ++kb) {
                FragH A, B;
#pragma unroll
                for (int v = 0; v < 8; ++v) A.u[v] = af[8 * kb + v];
#pragma unroll
                for (int v = 0; v < 8; ++v) B.u[v] = sV[m][16 * kb + (v & 3) + 8 * (v >> 2) + 4 * hi][l15];
                acc.v = __builtin_amdgcn_wmma_f32_16x16x32_f16(
                    false, A.h, false, B.h, (short)0, acc.v, false, false);
            }
#pragma unroll
            for (int v = 0; v < 8; ++v)
                sY[wave][v + 8 * hi][16 * m + l15] = to_h16(acc.f[v] * Cm[v]);
        };

        // ---- build w fragments, consume them (m=0,1), then retire wf ----
        {
            uint32_t wf[32];   // fragment kb = &wf[8*kb]
#pragma unroll
            for (int c = 0; c < 8; ++c) {
#pragma unroll
                for (int u2 = 0; u2 < 4; ++u2) {
                    float r0 = rr[c * 8 + 2 * u2], r1 = rr[c * 8 + 2 * u2 + 1];
                    wf[c * 4 + u2] = pack2h(aw * r0 + bw, aw * r1 + bw);
                }
            }
            gemm1_one(wf, 0, H10);
            gemm1_one(wf, 1, H11);
        }
        // ---- build delta_tilde fragments (retiring rr), consume them (m=2..5) ----
        {
            uint32_t df[32];
#pragma unroll
            for (int c = 0; c < 8; ++c) {
                float4 qa = *(const float4*)&sSp[hi][c * 8];
                float4 qb = *(const float4*)&sSp[hi][c * 8 + 4];
                float qs[8] = { qa.x, qa.y, qa.z, qa.w, qb.x, qb.y, qb.z, qb.w };
#pragma unroll
                for (int u2 = 0; u2 < 4; ++u2) {
                    float r0 = rr[c * 8 + 2 * u2], r1 = rr[c * 8 + 2 * u2 + 1];
                    df[c * 4 + u2] = pack2h((r0 - coef * qs[2 * u2])     * invdt,
                                            (r1 - coef * qs[2 * u2 + 1]) * invdt);
                }
            }
            gemm1_one(df, 2, H00);
            gemm1_one(df, 3, H01);
            gemm1_one(df, 4, H10);
            gemm1_one(df, 5, H11);
        }
        asm volatile("s_wait_dscnt 0" ::: "memory");  // sY writes visible wave-wide

        // ---- GEMM2: out = Y(16x96) @ Ucat^T(96x128) + h00*P0 + h01*P1 ----
        FragH A2[3];
#pragma unroll
        for (int kb = 0; kb < 3; ++kb) {
#pragma unroll
            for (int v = 0; v < 8; ++v) {
                int pidx = 16 * kb + (v & 3) + 8 * (v >> 2) + 4 * hi;
                A2[kb].u[v] = *(const uint32_t*)&sY[wave][l15][2 * pidx];
            }
        }
#pragma unroll
        for (int nt = 0; nt < 8; ++nt) {
            AccF o;
#pragma unroll
            for (int v = 0; v < 8; ++v) o.f[v] = 0.f;
#pragma unroll
            for (int kb = 0; kb < 3; ++kb) {
                FragH B;
#pragma unroll
                for (int v = 0; v < 8; ++v) B.u[v] = sU[16 * kb + (v & 3) + 8 * (v >> 2) + 4 * hi][16 * nt + l15];
                o.v = __builtin_amdgcn_wmma_f32_16x16x32_f16(
                    false, A2[kb].h, false, B.h, (short)0, o.v, false, false);
            }
#pragma unroll
            for (int v = 0; v < 8; ++v) {
                float res = o.f[v] + H00[v] * P0v[nt] + H01[v] * P1v[nt];
                p.out[(size_t)(rowBase + v + 8 * hi) * DDIM + 16 * nt + l15] = res;
            }
        }
    }
}

extern "C" void kernel_launch(void* const* d_in, const int* in_sizes, int n_in,
                              void* d_out, int out_size, void* d_ws, size_t ws_size,
                              hipStream_t stream) {
    (void)in_sizes; (void)n_in; (void)out_size; (void)d_ws; (void)ws_size;
    Params p;
    p.r     = (const float*)d_in[0];
    p.spine = (const float*)d_in[1];
    p.P0    = (const float*)d_in[2];
    p.P1    = (const float*)d_in[3];
    for (int m = 0; m < 6; ++m) {
        p.U[m] = (const float*)d_in[4 + 2 * m];   // Pp0_U, Pp1_U, E0_U, E1_U, Ep0_U, Ep1_U
        p.V[m] = (const float*)d_in[5 + 2 * m];   // matching _V
    }
    p.out = (float*)d_out;
    hipLaunchKernelGGL(spine_wmma_kernel, dim3(GRID), dim3(BLOCK_T), 0, stream, p);
}
